// SequencedLSTMs_81758997447125
// MI455X (gfx1250) — matrix operand order repaired and checked by reading, not compiled, not run on hardware
//
#include <hip/hip_runtime.h>
#include <hip/hip_bf16.h>

// Persistent fused 2-layer LayerNorm-LSTM for MI455X (gfx1250, wave32, WMMA).
//   grid = 16 workgroups (one per 16 batch rows), block = 512 threads (16 waves)
//   Wh1 bf16 fragments pinned in VGPRs; [Wx2;Wh2] bf16 fragments in LDS (256 KB)
//   h1/h2 in LDS (bf16, A-operand), c1/c2 in registers, z scratch in LDS (f32)
//   Layer-2's Wh2 half (depends only on old h2) is issued BEFORE layer-1's
//   LayerNorm so those WMMAs co-execute with the LN VALU work.
//   One kernel launch does all T=3000 steps + final classifier.

#define Tn 3000
#define Bn 256
#define Hn 128
#define Gn 512           // 4*H
#define WGB 16           // batch rows per workgroup
#define NWG (Bn / WGB)   // 16 workgroups
#define NTHREADS 512
#define ZP 516           // z row pitch in dwords (512 + 4 pad)
#define HP 272           // h12 row pitch in halves (256 + 16 pad)
#define EPSf 1e-5f

typedef __attribute__((ext_vector_type(16))) __bf16 v16bf;
typedef __attribute__((ext_vector_type(8)))  __bf16 v8bf;
typedef __attribute__((ext_vector_type(8)))  float  v8f;

struct __align__(64) SharedMem {
  __bf16 w2frag[256 * 512];          // layer-2 B fragments, 262144 B
  float  zs[WGB * ZP];               // gate pre-activations, 33024 B
  __bf16 h12[WGB * HP];              // [h1 | h2] bf16, A operand, 8704 B
  float  wx1[Gn], b1[Gn], gg1[Gn], bg1[Gn];
  float  b2[Gn],  gg2[Gn], bg2[Gn];
  float  gc1[Hn], bc1[Hn], gc2[Hn], bc2[Hn];
  float  wcls[Hn];
  float  xst[WGB];
};

__device__ __forceinline__ float sigf(float x) { return 1.f / (1.f + __expf(-x)); }
__device__ __forceinline__ float tanh_fast(float x) { return 2.f * sigf(2.f * x) - 1.f; }

// A-operand (16x32 bf16) fragment load from LDS h12, contiguous K order:
// lanes 0-15 hold K = kb..kb+15; lanes 16-31 hold K = kb+16..kb+31.
__device__ __forceinline__ v16bf load_a(const __bf16* h12, int lane, int kbase) {
  int row = lane & 15;
  int kb  = kbase + ((lane >> 4) << 4);
  const __bf16* p = h12 + row * HP + kb;
  v8bf lo = *(const v8bf*)(p);
  v8bf hi = *(const v8bf*)(p + 8);
  return __builtin_shufflevector(lo, hi, 0,1,2,3,4,5,6,7,8,9,10,11,12,13,14,15);
}

// Per-row LayerNorm + gates + cell/hidden update.  Row m == wave index.
// Each lane owns 4 hidden channels j = 4*lane .. 4*lane+3.
__device__ __forceinline__ void gate_update(
    SharedMem& s, int m, int lane, float* c,
    const float* gg, const float* bg, const float* gc, const float* bc,
    int hcol0, float* hkeep)
{
  const float* zrow = &s.zs[m * ZP];
  float sum = 0.f, sq = 0.f;
  #pragma unroll
  for (int q = 0; q < 16; ++q) {           // strided: conflict-free LDS banks
    float v = zrow[lane + 32 * q];
    sum += v; sq += v * v;
  }
  #pragma unroll
  for (int off = 16; off >= 1; off >>= 1) {
    sum += __shfl_xor(sum, off, 32);
    sq  += __shfl_xor(sq,  off, 32);
  }
  float mu = sum * (1.f / 512.f);
  float rs = rsqrtf(sq * (1.f / 512.f) - mu * mu + EPSf);

  int j0 = lane * 4;
  float osig[4];
  float csum = 0.f, csq = 0.f;
  #pragma unroll
  for (int q = 0; q < 4; ++q) {
    int j = j0 + q;
    float zi = (zrow[j         ] - mu) * rs * gg[j         ] + bg[j         ];
    float zf = (zrow[Hn + j    ] - mu) * rs * gg[Hn + j    ] + bg[Hn + j    ];
    float zg = (zrow[2*Hn + j  ] - mu) * rs * gg[2*Hn + j  ] + bg[2*Hn + j  ];
    float zo = (zrow[3*Hn + j  ] - mu) * rs * gg[3*Hn + j  ] + bg[3*Hn + j  ];
    float cn = sigf(zf) * c[q] + sigf(zi) * tanh_fast(zg);
    c[q] = cn;
    csum += cn; csq += cn * cn;
    osig[q] = sigf(zo);
  }
  #pragma unroll
  for (int off = 16; off >= 1; off >>= 1) {
    csum += __shfl_xor(csum, off, 32);
    csq  += __shfl_xor(csq,  off, 32);
  }
  float cmu = csum * (1.f / 128.f);
  float crs = rsqrtf(csq * (1.f / 128.f) - cmu * cmu + EPSf);
  #pragma unroll
  for (int q = 0; q < 4; ++q) {
    int j = j0 + q;
    float hv = osig[q] * tanh_fast((c[q] - cmu) * crs * gc[j] + bc[j]);
    hkeep[q] = hv;
    s.h12[m * HP + hcol0 + j] = (__bf16)hv;
  }
}

__global__ __launch_bounds__(NTHREADS, 1)
void lnlstm2_persistent_kernel(
    const float* __restrict__ x,
    const float* __restrict__ h01,  const float* __restrict__ h02,
    const float* __restrict__ Wx1,  const float* __restrict__ Wh1,
    const float* __restrict__ b1,   const float* __restrict__ gg1,
    const float* __restrict__ bg1,  const float* __restrict__ gc1,
    const float* __restrict__ bc1,  const float* __restrict__ Wx2,
    const float* __restrict__ Wh2,  const float* __restrict__ b2,
    const float* __restrict__ gg2,  const float* __restrict__ bg2,
    const float* __restrict__ gc2,  const float* __restrict__ bc2,
    const float* __restrict__ wcls, const float* __restrict__ bcls,
    float* __restrict__ out)
{
  __shared__ SharedMem s;
  const int tid  = threadIdx.x;
  const int wave = tid >> 5;
  const int lane = tid & 31;
  const int wg   = blockIdx.x;
  const int row0 = wg * WGB;

  // ---- stage constant vectors into LDS ----
  for (int i = tid; i < Gn; i += NTHREADS) {
    s.wx1[i] = Wx1[i]; s.b1[i] = b1[i]; s.gg1[i] = gg1[i]; s.bg1[i] = bg1[i];
    s.b2[i]  = b2[i];  s.gg2[i] = gg2[i]; s.bg2[i] = bg2[i];
  }
  if (tid < Hn) {
    s.gc1[tid] = gc1[tid]; s.bc1[tid] = bc1[tid];
    s.gc2[tid] = gc2[tid]; s.bc2[tid] = bc2[tid];
    s.wcls[tid] = wcls[tid];
  }

  // ---- init h12 = [h0_1 | h0_2] (bf16) for this workgroup's rows ----
  for (int i = tid; i < WGB * 256; i += NTHREADS) {
    int m = i >> 8, k = i & 255;
    float v = (k < Hn) ? h01[(row0 + m) * Hn + k]
                       : h02[(row0 + m) * Hn + (k - Hn)];
    s.h12[m * HP + k] = (__bf16)v;
  }

  // ---- build layer-2 B fragments in LDS: W2cat = [Wx2 ; Wh2] (512 x 256) ----
  // fragment f = ntile*8 + kstep; per-lane 16 contiguous halves (32 B):
  // lane L: n = ntile*16 + (L&15), K = kstep*32 + (L<16 ? 0 : 16) + e
  for (int idx = tid; idx < 256 * 32; idx += NTHREADS) {
    int f = idx >> 5, l = idx & 31;
    int ntile = f >> 3, ks = f & 7;
    int n  = ntile * 16 + (l & 15);
    int kb = ks * 32 + ((l < 16) ? 0 : 16);
    const float* src = (kb < Hn) ? (Wx2 + n * Hn + kb)
                                 : (Wh2 + n * Hn + (kb - Hn));
    __bf16* dst = &s.w2frag[f * 512 + l * 16];
    #pragma unroll
    for (int e = 0; e < 16; ++e) dst[e] = (__bf16)src[e];
  }

  // ---- pin this wave's Wh1 B fragments in VGPRs (loop-invariant) ----
  v16bf wh1f[2][4];
  #pragma unroll
  for (int tt = 0; tt < 2; ++tt) {
    #pragma unroll
    for (int ks = 0; ks < 4; ++ks) {
      int n  = (wave * 2 + tt) * 16 + (lane & 15);
      int kb = ks * 32 + ((lane < 16) ? 0 : 16);
      const float* src = Wh1 + n * Hn + kb;
      v16bf v;
      #pragma unroll
      for (int e = 0; e < 16; ++e) v[e] = (__bf16)src[e];
      wh1f[tt][ks] = v;
    }
  }

  if (tid < WGB) s.xst[tid] = x[(row0 + tid) * Tn + 0];

  float c1[4] = {0.f, 0.f, 0.f, 0.f};
  float c2[4] = {0.f, 0.f, 0.f, 0.f};
  float h1keep[4];
  float h2reg[4] = {0.f, 0.f, 0.f, 0.f};

  __syncthreads();

  for (int t = 0; t < Tn; ++t) {
    // ---- layer 1 GEMM: z1 = h1 @ Wh1^T  (per wave: 2 N-tiles, K=128) ----
    v8f acc1[2];
    #pragma unroll
    for (int tt = 0; tt < 2; ++tt) {
      v8f a = {0.f, 0.f, 0.f, 0.f, 0.f, 0.f, 0.f, 0.f};
      #pragma unroll
      for (int ks = 0; ks < 4; ++ks) {
        v16bf av = load_a(s.h12, lane, ks * 32);
        a = __builtin_amdgcn_wmma_f32_16x16x32_bf16(
                false, av, false, wh1f[tt][ks], (short)0, a, false, false);
      }
      acc1[tt] = a;
    }

    // ---- layer 2 partial GEMM: the Wh2 half (old h2, ks=4..7).  These 8
    //      WMMAs have no dependence on layer-1's output, so they overlap
    //      (XDL co-execution) with the z1 stores and layer-1 LayerNorm. ----
    v8f acc2[2];
    #pragma unroll
    for (int tt = 0; tt < 2; ++tt) {
      v8f a = {0.f, 0.f, 0.f, 0.f, 0.f, 0.f, 0.f, 0.f};
      #pragma unroll
      for (int ks = 4; ks < 8; ++ks) {
        v16bf av = load_a(s.h12, lane, ks * 32);     // old h2 columns
        v16bf bv = *(const v16bf*)&s.w2frag[((wave * 2 + tt) * 8 + ks) * 512 + lane * 16];
        a = __builtin_amdgcn_wmma_f32_16x16x32_bf16(
                false, av, false, bv, (short)0, a, false, false);
      }
      acc2[tt] = a;
    }

    // z1 += x_t * Wx1 + b1 ; write to LDS (C layout: lane owns col n, 8 rows)
    #pragma unroll
    for (int tt = 0; tt < 2; ++tt) {
      int n   = (wave * 2 + tt) * 16 + (lane & 15);
      float wxn = s.wx1[n], bn = s.b1[n];
      int mh = (lane >> 4) * 8;
      #pragma unroll
      for (int r = 0; r < 8; ++r)
        s.zs[(mh + r) * ZP + n] = acc1[tt][r] + s.xst[mh + r] * wxn + bn;
    }
    __syncthreads();

    // ---- layer 1 LN + gates; writes new h1 (bf16) into h12 cols 0..127 ----
    gate_update(s, wave, lane, c1, s.gg1, s.bg1, s.gc1, s.bc1, 0, h1keep);
    __syncthreads();

    // ---- finish layer 2 GEMM: the Wx2 half (new h1, ks=0..3) ----
    #pragma unroll
    for (int tt = 0; tt < 2; ++tt) {
      v8f a = acc2[tt];
      #pragma unroll
      for (int ks = 0; ks < 4; ++ks) {
        v16bf av = load_a(s.h12, lane, ks * 32);     // new h1 columns
        v16bf bv = *(const v16bf*)&s.w2frag[((wave * 2 + tt) * 8 + ks) * 512 + lane * 16];
        a = __builtin_amdgcn_wmma_f32_16x16x32_bf16(
                false, av, false, bv, (short)0, a, false, false);
      }
      acc2[tt] = a;
    }
    #pragma unroll
    for (int tt = 0; tt < 2; ++tt) {
      int n  = (wave * 2 + tt) * 16 + (lane & 15);
      float bn = s.b2[n];
      int mh = (lane >> 4) * 8;
      #pragma unroll
      for (int r = 0; r < 8; ++r)
        s.zs[(mh + r) * ZP + n] = acc2[tt][r] + bn;
    }
    __syncthreads();

    // ---- layer 2 LN + gates; writes new h2 into h12 cols 128..255 ----
    gate_update(s, wave, lane, c2, s.gg2, s.bg2, s.gc2, s.bc2, Hn, h2reg);
    if (tid < WGB && t + 1 < Tn)
      s.xst[tid] = x[(row0 + tid) * Tn + (t + 1)];
    __syncthreads();
  }

  // ---- classifier: logit[m] = h2[m,:] . wcls + bcls ; sigmoid ----
  float part = 0.f;
  #pragma unroll
  for (int q = 0; q < 4; ++q) part += h2reg[q] * s.wcls[lane * 4 + q];
  #pragma unroll
  for (int off = 16; off >= 1; off >>= 1) part += __shfl_xor(part, off, 32);
  if (lane == 0)
    out[row0 + wave] = 1.f / (1.f + __expf(-(part + bcls[0])));
}

extern "C" void kernel_launch(void* const* d_in, const int* in_sizes, int n_in,
                              void* d_out, int out_size, void* d_ws, size_t ws_size,
                              hipStream_t stream) {
  (void)in_sizes; (void)n_in; (void)out_size; (void)d_ws; (void)ws_size;
  const float* x    = (const float*)d_in[0];
  const float* h01  = (const float*)d_in[1];
  const float* h02  = (const float*)d_in[2];
  const float* Wx1  = (const float*)d_in[3];
  const float* Wh1  = (const float*)d_in[4];
  const float* b1   = (const float*)d_in[5];
  const float* gg1  = (const float*)d_in[6];
  const float* bg1  = (const float*)d_in[7];
  const float* gc1  = (const float*)d_in[8];
  const float* bc1  = (const float*)d_in[9];
  const float* Wx2  = (const float*)d_in[10];
  const float* Wh2  = (const float*)d_in[11];
  const float* b2   = (const float*)d_in[12];
  const float* gg2  = (const float*)d_in[13];
  const float* bg2  = (const float*)d_in[14];
  const float* gc2  = (const float*)d_in[15];
  const float* bc2  = (const float*)d_in[16];
  const float* wcls = (const float*)d_in[17];
  const float* bcls = (const float*)d_in[18];
  float* out = (float*)d_out;

  lnlstm2_persistent_kernel<<<dim3(NWG), dim3(NTHREADS), 0, stream>>>(
      x, h01, h02, Wx1, Wh1, b1, gg1, bg1, gc1, bc1,
      Wx2, Wh2, b2, gg2, bg2, gc2, bc2, wcls, bcls, out);
}
